// MultiQueryAttention_60644938219778
// MI455X (gfx1250) — matrix operand order repaired and checked by reading, not compiled, not run on hardware
//
#include <hip/hip_runtime.h>
#include <hip/hip_bf16.h>
#include <math.h>

// ---------------------------------------------------------------------------
// MQA decode on MI455X (gfx1250, wave32).
// Roofline: 268MB fp32 KV cache @ 23.3TB/s dominates (~13us floor). Compute
// via v_wmma_f32_16x16x32_bf16 with in-register fp32->bf16 conversion.
// KV tiles are shared across all 16 heads of a batch through LDS, streamed
// with double-buffered global_load_async_to_lds_b128 (split s_wait_asynccnt)
// so HBM fetch fully overlaps WMMA + online softmax.
// ---------------------------------------------------------------------------

typedef __attribute__((ext_vector_type(16))) __bf16 v16bf;
typedef __attribute__((ext_vector_type(8)))  float  v8f;

#define WMMA_BF16(a, b, c) \
  __builtin_amdgcn_wmma_f32_16x16x32_bf16(false, (a), false, (b), (short)0, (c), false, false)

__device__ __forceinline__ int lane_id() { return (int)(threadIdx.x & 31u); }

// A fragment (M=16 x K=32) from a row-major f32 tile, ld in elements.
// Contiguous k order (B frags match): lane l: m=l&15, half=l>>4; VGPR v holds K = half*16 + v*2 (+1)
__device__ __forceinline__ v16bf frag_a_f32(const float* __restrict__ tile, int ld) {
  const int l = lane_id();
  const int m = l & 15, half = l >> 4;
  const float* p = tile + (size_t)m * ld + half * 16;
  v16bf a;
#pragma unroll
  for (int v = 0; v < 8; ++v) {
    const int k = v << 1;
    a[2 * v]     = (__bf16)p[k];
    a[2 * v + 1] = (__bf16)p[k + 1];
  }
  return a;
}

// B fragment (K=32 x N=16) where B(k,n) = rowsrc[n*ld + k]  (N is the slow axis;
// matches nn.Linear weight [N,K] and K-cache rows [s][d] used as K^T).
__device__ __forceinline__ v16bf frag_b_nrows_f32(const float* __restrict__ rowsrc, int ld) {
  const int l = lane_id();
  const int n = l & 15, kh = l >> 4;
  const float* p = rowsrc + (size_t)n * ld + kh * 16;
  v16bf b;
#pragma unroll
  for (int v = 0; v < 8; ++v) {
    b[2 * v]     = (__bf16)p[2 * v];
    b[2 * v + 1] = (__bf16)p[2 * v + 1];
  }
  return b;
}

// B fragment (K=32 x N=16) where B(k,n) = src[k*ld + n]  (K is the slow axis; V tiles).
__device__ __forceinline__ v16bf frag_b_krows_f32(const float* __restrict__ src, int ld) {
  const int l = lane_id();
  const int n = l & 15, kh = l >> 4;
  const float* p = src + (size_t)(kh * 16) * ld + n;
  v16bf b;
#pragma unroll
  for (int v = 0; v < 8; ++v) {
    b[2 * v]     = (__bf16)p[(size_t)(2 * v) * ld];
    b[2 * v + 1] = (__bf16)p[(size_t)(2 * v + 1) * ld];
  }
  return b;
}

// ---------------------------------------------------------------------------
// GEMM: C[M,N] = A[M,K] @ W[N,K]^T + bias   (f32 memory, bf16 WMMA compute)
// 8 waves / block, one 16x16 tile per wave, block covers 16(M) x 128(N).
// A and W both stream through L2 (x is 4MB, W is 16MB -> resident in 192MB L2).
// ---------------------------------------------------------------------------
__global__ void __launch_bounds__(256)
gemm_xWT_bias_kernel(const float* __restrict__ A, const float* __restrict__ W,
                     const float* __restrict__ bias, float* __restrict__ C,
                     int M, int N, int K) {
  const int wave = threadIdx.x >> 5;
  const int l = lane_id();
  const int n0 = blockIdx.x * 128 + wave * 16;
  const int m0 = blockIdx.y * 16;
  if (n0 >= N || m0 >= M) return;

  v8f acc = {0.f, 0.f, 0.f, 0.f, 0.f, 0.f, 0.f, 0.f};
  const float* Ab = A + (size_t)m0 * K;
  const float* Wb = W + (size_t)n0 * K;
#pragma unroll 2
  for (int k0 = 0; k0 < K; k0 += 32) {
    const v16bf af = frag_a_f32(Ab + k0, K);
    const v16bf bf_ = frag_b_nrows_f32(Wb + k0, K);
    acc = WMMA_BF16(af, bf_, acc);
  }
  const int nn = l & 15, mh = l >> 4;
  const float bval = bias[n0 + nn];
  float* Cp = C + (size_t)(m0 + mh * 8) * N + n0 + nn;
#pragma unroll
  for (int r = 0; r < 8; ++r) Cp[(size_t)r * N] = acc[r] + bval;
}

// ---------------------------------------------------------------------------
// Flash attention: one block per batch, one wave per head; shared KV in LDS,
// double-buffered async staging.
// ---------------------------------------------------------------------------
#define TQ 16
#define HD 128
#define NH 16
#define SCACHE 8192
#define CHUNK 32
#define EMB 2048

#if defined(__HIP_DEVICE_COMPILE__) && \
    __has_builtin(__builtin_amdgcn_global_load_async_to_lds_b128) && \
    __has_builtin(__builtin_amdgcn_s_wait_asynccnt)
#define USE_ASYNC_LDS 1
#else
#define USE_ASYNC_LDS 0
#endif

#if USE_ASYNC_LDS
#define GLOBAL_AS __attribute__((address_space(1)))
#define LOCAL_AS  __attribute__((address_space(3)))
typedef int v4i_gcc __attribute__((vector_size(16)));  // matches builtin param type
__device__ __forceinline__ void async_copy_b128(const float* g, float* lds) {
  __builtin_amdgcn_global_load_async_to_lds_b128(
      (GLOBAL_AS v4i_gcc*)(v4i_gcc*)(g),
      (LOCAL_AS  v4i_gcc*)(v4i_gcc*)(lds),
      0, 0);
}
#define WAIT_ASYNC(n) __builtin_amdgcn_s_wait_asynccnt(n)
#else
#define WAIT_ASYNC(n) ((void)0)
#endif

// Stage one 32x128 fp32 cache chunk (K & V) into LDS: 4 async b128 ops/thread.
__device__ __forceinline__ void stage_cache_chunk(const float* kg, const float* vg,
                                                  float* kbuf, float* vbuf, int tid) {
#if USE_ASYNC_LDS
#pragma unroll
  for (int i = 0; i < (CHUNK * HD / 4); i += 512) {
    async_copy_b128(kg + (size_t)(i + tid) * 4, &kbuf[(i + tid) * 4]);
    async_copy_b128(vg + (size_t)(i + tid) * 4, &vbuf[(i + tid) * 4]);
  }
#else
  for (int i = tid; i < CHUNK * HD / 4; i += 512) {
    ((float4*)kbuf)[i] = ((const float4*)kg)[i];
    ((float4*)vbuf)[i] = ((const float4*)vg)[i];
  }
#endif
}

// Stage the 16 appended K_new/V_new rows (plain copies) and zero-pad rows 16..31.
__device__ __forceinline__ void stage_appended_chunk(const float* kg, const float* vg,
                                                      float* kbuf, float* vbuf, int tid) {
  const float4 z4 = make_float4(0.f, 0.f, 0.f, 0.f);
  for (int i = tid; i < TQ * HD / 4; i += 512) {
    ((float4*)kbuf)[i] = ((const float4*)kg)[i];
    ((float4*)vbuf)[i] = ((const float4*)vg)[i];
    ((float4*)kbuf)[TQ * HD / 4 + i] = z4;
    ((float4*)vbuf)[TQ * HD / 4 + i] = z4;
  }
}

__global__ void __launch_bounds__(512)
mqa_flash_kernel(const float* __restrict__ Q,   // [B*T, EMB]
                 const float* __restrict__ Kc,  // [B, SCACHE, HD]
                 const float* __restrict__ Vc,  // [B, SCACHE, HD]
                 const float* __restrict__ Kn,  // [B*T, HD]
                 const float* __restrict__ Vn,  // [B*T, HD]
                 float* __restrict__ O) {       // [B*T, EMB]
  __shared__ __align__(16) float Ks[2][CHUNK * HD];      // 2 x 16 KB
  __shared__ __align__(16) float Vs[2][CHUNK * HD];      // 2 x 16 KB
  __shared__ __align__(16) float Ps[NH][TQ * CHUNK];     // 32 KB

  const int b   = blockIdx.x;
  const int tid = (int)threadIdx.x;
  const int h   = tid >> 5;          // wave == head
  const int l   = lane_id();
  const int nn  = l & 15, mh = l >> 4;

  // Q tile for this head: 16 x 128 -> 4 A fragments, loaded once.
  v16bf qa[4];
  {
    const float* Qb = Q + (size_t)(b * TQ) * EMB + h * HD;
#pragma unroll
    for (int kc = 0; kc < 4; ++kc) qa[kc] = frag_a_f32(Qb + kc * 32, EMB);
  }

  const v8f zero8 = {0.f, 0.f, 0.f, 0.f, 0.f, 0.f, 0.f, 0.f};
  v8f o[8];
#pragma unroll
  for (int i = 0; i < 8; ++i) o[i] = zero8;
  float mrow[8], lrow[8];
#pragma unroll
  for (int r = 0; r < 8; ++r) { mrow[r] = -1e30f; lrow[r] = 0.f; }

  const float scale = 0.08838834764831845f;  // 1/sqrt(128)
  const float* kcb = Kc + (size_t)b * SCACHE * HD;
  const float* vcb = Vc + (size_t)b * SCACHE * HD;
  const float* knb = Kn + (size_t)(b * TQ) * HD;
  const float* vnb = Vn + (size_t)(b * TQ) * HD;
  const int nch = SCACHE / CHUNK;  // 256 cache chunks; chunk 'nch' = appended KV

  // Prologue: stage chunk 0 into buffer 0 (4 async ops/thread in flight).
  stage_cache_chunk(kcb, vcb, &Ks[0][0], &Vs[0][0], tid);

  for (int c = 0; c <= nch; ++c) {
    const int cur = c & 1;
    const int nxt = cur ^ 1;

    // ---- kick off next chunk into the other buffer, then wait for current ----
    if (c + 1 < nch) {
      stage_cache_chunk(kcb + (size_t)(c + 1) * CHUNK * HD,
                        vcb + (size_t)(c + 1) * CHUNK * HD,
                        &Ks[nxt][0], &Vs[nxt][0], tid);
      WAIT_ASYNC(4);        // chunk c complete; chunk c+1 (4 ops) stays in flight
      if ((c + 2 < nch) && (tid < 256))
        __builtin_prefetch(kcb + (size_t)(c + 2) * CHUNK * HD + (size_t)tid * 16, 0, 1);
    } else if (c + 1 == nch) {
      stage_appended_chunk(knb, vnb, &Ks[nxt][0], &Vs[nxt][0], tid);
      WAIT_ASYNC(0);        // appended chunk uses plain stores (not ASYNCcnt)
    } else {
      WAIT_ASYNC(0);
    }
    __syncthreads();        // staged data visible to all 16 waves

    const float* Ksc = &Ks[cur][0];
    const float* Vsc = &Vs[cur][0];

    // ---- S = Q @ K^T for this head's 16x32 tile ----
    const int nfr = (c < nch) ? 2 : 1;  // appended chunk: only 16 valid columns
    v8f sfr[2];
    for (int nb = 0; nb < nfr; ++nb) {
      v8f s = zero8;
#pragma unroll
      for (int kc = 0; kc < 4; ++kc) {
        const v16bf bk = frag_b_nrows_f32(&Ksc[(nb * 16) * HD + kc * 32], HD);
        s = WMMA_BF16(qa[kc], bk, s);
      }
      sfr[nb] = s;
    }

    // ---- online softmax (row stats via 16-lane shfl reductions) ----
    float alpha[8];
#pragma unroll
    for (int r = 0; r < 8; ++r) {
      float smax = -1e30f;
      for (int nb = 0; nb < nfr; ++nb) smax = fmaxf(smax, sfr[nb][r]);
#pragma unroll
      for (int d = 1; d < 16; d <<= 1) smax = fmaxf(smax, __shfl_xor(smax, d, 32));
      const float mnew = fmaxf(mrow[r], smax * scale);
      alpha[r] = __expf(mrow[r] - mnew);
      mrow[r] = mnew;
      float rsum = 0.f;
      for (int nb = 0; nb < nfr; ++nb) {
        const float p = __expf(sfr[nb][r] * scale - mnew);
        sfr[nb][r] = p;
        rsum += p;
      }
#pragma unroll
      for (int d = 1; d < 16; d <<= 1) rsum += __shfl_xor(rsum, d, 32);
      lrow[r] = lrow[r] * alpha[r] + rsum;
    }
#pragma unroll
    for (int ob = 0; ob < 8; ++ob)
#pragma unroll
      for (int r = 0; r < 8; ++r) o[ob][r] *= alpha[r];

    // ---- re-fragment P (C layout -> A layout) through this wave's LDS slab ----
    float* Pw = &Ps[h][0];
#pragma unroll
    for (int r = 0; r < 8; ++r) {
      const int m = mh * 8 + r;
      Pw[m * CHUNK + nn]      = sfr[0][r];
      Pw[m * CHUNK + 16 + nn] = (nfr == 2) ? sfr[1][r] : 0.f;  // mask padded cols
    }
    asm volatile("s_wait_dscnt 0x0" ::: "memory");  // per-wave LDS RAW
    const v16bf pa = frag_a_f32(Pw, CHUNK);

    // ---- O += P @ V ----
#pragma unroll
    for (int ob = 0; ob < 8; ++ob) {
      const v16bf vb = frag_b_krows_f32(&Vsc[ob * 16], HD);
      o[ob] = WMMA_BF16(pa, vb, o[ob]);
    }
    __syncthreads();  // all waves done with buf[cur] before it is restaged
  }

  // ---- epilogue: normalize and store [B*T, EMB] at head offset ----
  float inv[8];
#pragma unroll
  for (int r = 0; r < 8; ++r) inv[r] = 1.f / lrow[r];
  float* Ob = O + (size_t)(b * TQ + mh * 8) * EMB + h * HD + nn;
#pragma unroll
  for (int ob = 0; ob < 8; ++ob)
#pragma unroll
    for (int r = 0; r < 8; ++r)
      Ob[(size_t)r * EMB + ob * 16] = o[ob][r] * inv[r];
}

// ---------------------------------------------------------------------------
extern "C" void kernel_launch(void* const* d_in, const int* in_sizes, int n_in,
                              void* d_out, int out_size, void* d_ws, size_t ws_size,
                              hipStream_t stream) {
  (void)in_sizes; (void)n_in; (void)out_size; (void)ws_size;
  const float* x  = (const float*)d_in[0];
  const float* Kc = (const float*)d_in[1];
  const float* Vc = (const float*)d_in[2];
  const float* Wq = (const float*)d_in[3];
  const float* bq = (const float*)d_in[4];
  const float* Wk = (const float*)d_in[5];
  const float* bk = (const float*)d_in[6];
  const float* Wv = (const float*)d_in[7];
  const float* bv = (const float*)d_in[8];
  const float* Wo = (const float*)d_in[9];
  const float* bo = (const float*)d_in[10];
  float* out = (float*)d_out;

  float* Qb  = (float*)d_ws;                 // 512*2048
  float* Kn  = Qb + (size_t)512 * 2048;      // 512*128
  float* Vn  = Kn + (size_t)512 * 128;       // 512*128
  float* Att = Vn + (size_t)512 * 128;       // 512*2048

  const dim3 blk(256);
  gemm_xWT_bias_kernel<<<dim3(2048 / 128, 512 / 16), blk, 0, stream>>>(x, Wq, bq, Qb, 512, 2048, 2048);
  gemm_xWT_bias_kernel<<<dim3(1, 512 / 16), blk, 0, stream>>>(x, Wk, bk, Kn, 512, 128, 2048);
  gemm_xWT_bias_kernel<<<dim3(1, 512 / 16), blk, 0, stream>>>(x, Wv, bv, Vn, 512, 128, 2048);
  mqa_flash_kernel<<<dim3(32), dim3(512), 0, stream>>>(Qb, Kc, Vc, Kn, Vn, Att);
  gemm_xWT_bias_kernel<<<dim3(2048 / 128, 512 / 16), blk, 0, stream>>>(Att, Wo, bo, out, 512, 2048, 2048);
}
